// VectorQuantizer_26250840113735
// MI455X (gfx1250) — matrix operand order repaired and checked by reading, not compiled, not run on hardware
//
#include <hip/hip_runtime.h>
#include <stdint.h>

typedef __attribute__((ext_vector_type(4)))  _Float16 v4h;
typedef __attribute__((ext_vector_type(8)))  _Float16 v8h;
typedef __attribute__((ext_vector_type(16))) _Float16 v16h;
typedef __attribute__((ext_vector_type(8)))  float    v8f;

#define D       256
#define KCODES  1024
#define NROW    32768
#define MT      64            // z rows per block
#define NT      32            // codes per chunk
#define STRIDE  264           // halves; 132 dwords -> conflict-free b128 row reads
#define NCHUNK  (KCODES / NT) // 32
#define NBLK    (NROW / MT)   // 512

// Issue the two LDS b128 loads for one 32x16 f16 B fragment (no wait).
#define ISSUE_B(lo, hi, addr0)                                   \
  asm volatile("ds_load_b128 %0, %2\n\t"                         \
               "ds_load_b128 %1, %3"                             \
               : "=&v"(lo), "=&v"(hi)                            \
               : "v"(addr0), "v"((addr0) + 32)                   \
               : "memory")

// Wait until the fragment (lo,hi) is guaranteed landed, tying the registers
// through the asm so the consuming WMMA cannot be scheduled above the wait
// and so the two pipeline buffers cannot be register-coalesced.
#define WAIT_B2(lo, hi) \
  asm volatile("s_wait_dscnt 0x2" : "+v"(lo), "+v"(hi) :: "memory")
#define WAIT_B0(lo, hi) \
  asm volatile("s_wait_dscnt 0x0" : "+v"(lo), "+v"(hi) :: "memory")

__global__ __launch_bounds__(256)
void vq_main(const float* __restrict__ z, const float* __restrict__ emb,
             float* __restrict__ outq, float* __restrict__ partial) {
  __shared__ __align__(16) _Float16 Zt[MT * STRIDE];   // 33792 B
  __shared__ __align__(16) _Float16 Et[NT * STRIDE];   // 16896 B
  __shared__ float eN[NT];
  __shared__ float candD[2][MT];
  __shared__ int   candN[2][MT];
  __shared__ int   bestRow[MT];
  __shared__ float red[256];

  const int  t    = threadIdx.x;
  const int  lane = t & 31;
  const int  w    = t >> 5;
  const int  m0   = (w & 3) * 16;   // row sub-tile
  const int  c0   = (w >> 2) * 16;  // col sub-tile within chunk
  const int  bid  = blockIdx.x;
  const long row0 = (long)bid * MT;

  // ---- stage Z tile: f32 global -> f16 LDS (coalesced float4 reads) ----
  for (int u = t; u < MT * (D / 4); u += 256) {
    int r = u >> 6, c4 = u & 63;
    float4 v = ((const float4*)(z + (row0 + r) * D))[c4];
    v4h h = { (_Float16)v.x, (_Float16)v.y, (_Float16)v.z, (_Float16)v.w };
    *(v4h*)&Zt[r * STRIDE + c4 * 4] = h;
  }
  __syncthreads();

  const int mrow = m0 + (lane & 15);
  const int hi   = lane >> 4;

  // ---- hoist all 8 A-fragments (16x32 f16 each) into registers:
  //      they depend only on k-step, so load once and reuse for all 32 chunks.
  v16h areg[8];
#pragma unroll
  for (int ks = 0; ks < 8; ++ks) {
    const int k0 = ks * 32;
    v8h alo = *(const v8h*)&Zt[mrow * STRIDE + k0 + hi * 8];
    v8h ahi = *(const v8h*)&Zt[mrow * STRIDE + k0 + 16 + hi * 8];
    areg[ks] = __builtin_shufflevector(alo, ahi,
                 0, 1, 2, 3, 4, 5, 6, 7, 8, 9, 10, 11, 12, 13, 14, 15);
  }

  float bestD[8];
  int   bestN[8];
#pragma unroll
  for (int r = 0; r < 8; ++r) { bestD[r] = 3.402823466e38f; bestN[r] = 0; }

  // Per-lane base LDS byte address of this wave's B fragment (k0 = 0).
  const unsigned bbase = (unsigned)(size_t)(&Et[0])
      + (unsigned)(((c0 + (lane & 15)) * STRIDE) * 2 + hi * 16);

  for (int ch = 0; ch < NCHUNK; ++ch) {
    __syncthreads();  // previous chunk's Et fully consumed

    // ---- stage E chunk: f32 -> f16 LDS, f32 norms (8 threads per code) ----
    {
      int cl = t >> 3;   // local code 0..31
      int p  = t & 7;
      const float4* esrc = (const float4*)(emb + (size_t)(ch * NT + cl) * D);
      float nrm = 0.f;
#pragma unroll
      for (int j = 0; j < 8; ++j) {
        float4 v = esrc[p * 8 + j];
        nrm += v.x * v.x + v.y * v.y + v.z * v.z + v.w * v.w;
        v4h h = { (_Float16)v.x, (_Float16)v.y, (_Float16)v.z, (_Float16)v.w };
        *(v4h*)&Et[cl * STRIDE + (p * 8 + j) * 4] = h;
      }
      nrm += __shfl_xor(nrm, 1, 32);
      nrm += __shfl_xor(nrm, 2, 32);
      nrm += __shfl_xor(nrm, 4, 32);
      if (p == 0) eN[cl] = nrm;
    }
    __syncthreads();

    // ---- software-pipelined B b128 loads + WMMA over k ----
    v8f acc = {0.f, 0.f, 0.f, 0.f, 0.f, 0.f, 0.f, 0.f};
    v8h blo[2], bhi[2];
    ISSUE_B(blo[0], bhi[0], bbase);          // prologue: B(0)
#pragma unroll
    for (int ks = 0; ks < 8; ++ks) {
      const int cur = ks & 1;
      if (ks < 7) {
        ISSUE_B(blo[cur ^ 1], bhi[cur ^ 1], bbase + (unsigned)(ks + 1) * 64u);
        // The 2 newest DS ops are the pair just issued; in-order retirement
        // means dscnt<=2 guarantees B(ks) has landed.
        WAIT_B2(blo[cur], bhi[cur]);
      } else {
        WAIT_B0(blo[cur], bhi[cur]);
      }
      v16h b = __builtin_shufflevector(blo[cur], bhi[cur],
                 0, 1, 2, 3, 4, 5, 6, 7, 8, 9, 10, 11, 12, 13, 14, 15);
      acc = __builtin_amdgcn_wmma_f32_16x16x32_f16(
              false, areg[ks], false, b, (short)0, acc, false, false);
    }

    // ---- fused argmin epilogue: dist = ||e||^2 - 2 z.e (||z||^2 constant) ----
    const int   nloc = c0 + (lane & 15);
    const float en   = eN[nloc];
    const int   ng   = ch * NT + nloc;
#pragma unroll
    for (int r = 0; r < 8; ++r) {
      float dist = __builtin_fmaf(-2.f, acc[r], en);
      bool  tk   = dist < bestD[r];       // increasing n order -> '<' keeps first min
      bestD[r] = tk ? dist : bestD[r];
      bestN[r] = tk ? ng   : bestN[r];
    }
  }

  // ---- cross-lane reduce within 16-lane halves (rows m0+r and m0+r+8) ----
#pragma unroll
  for (int r = 0; r < 8; ++r) {
    float d = bestD[r];
    int   i = bestN[r];
#pragma unroll
    for (int off = 1; off < 16; off <<= 1) {
      float od = __shfl_xor(d, off, 32);
      int   oi = __shfl_xor(i, off, 32);
      bool  tk = (od < d) || (od == d && oi < i);  // argmin tie -> smaller index
      d = tk ? od : d;
      i = tk ? oi : i;
    }
    if ((lane & 15) == 0) {
      int rr = m0 + r + (lane >> 4) * 8;
      candD[c0 >> 4][rr] = d;
      candN[c0 >> 4][rr] = i;
    }
  }
  __syncthreads();
  if (t < MT) {  // merge the two wave-column groups
    float d0 = candD[0][t], d1 = candD[1][t];
    int   n0 = candN[0][t], n1 = candN[1][t];
    bool  tk = (d1 < d0) || (d1 == d0 && n1 < n0);
    bestRow[t] = tk ? n1 : n0;
  }
  __syncthreads();

  // ---- gather f32 codebook rows -> output, fused loss partial ----
  float lsum = 0.f;
  for (int u = t; u < MT * (D / 4); u += 256) {
    int r = u >> 6, c4 = u & 63;
    int idx = bestRow[r];
    float4 e4 = ((const float4*)(emb + (size_t)idx * D))[c4];
    float4 zv = ((const float4*)(z + (row0 + r) * D))[c4];
    ((float4*)(outq + (row0 + r) * D))[c4] = e4;  // straight-through == emb[idx]
    float dx = e4.x - zv.x, dy = e4.y - zv.y;
    float dz = e4.z - zv.z, dw = e4.w - zv.w;
    lsum += dx * dx + dy * dy + dz * dz + dw * dw;
  }
  red[t] = lsum;
  __syncthreads();
  for (int s = 128; s > 0; s >>= 1) {  // fixed-order tree -> deterministic
    if (t < s) red[t] += red[t + s];
    __syncthreads();
  }
  if (t == 0) partial[bid] = red[0];
}

__global__ __launch_bounds__(256)
void vq_loss(const float* __restrict__ partial, float* __restrict__ out) {
  __shared__ float red[256];
  int t = threadIdx.x;
  red[t] = partial[t] + partial[t + 256];
  __syncthreads();
  for (int s = 128; s > 0; s >>= 1) {
    if (t < s) red[t] += red[t + s];
    __syncthreads();
  }
  // loss = q_latent + 0.25*e_latent, both numerically mean((q - z)^2)
  if (t == 0) out[NROW * D] = 1.25f * red[0] / (float)(NROW * D);
}

extern "C" void kernel_launch(void* const* d_in, const int* in_sizes, int n_in,
                              void* d_out, int out_size, void* d_ws, size_t ws_size,
                              hipStream_t stream) {
  (void)in_sizes; (void)n_in; (void)out_size; (void)ws_size;
  const float* z   = (const float*)d_in[0];
  const float* emb = (const float*)d_in[1];
  float* outq    = (float*)d_out;
  float* partial = (float*)d_ws;  // NBLK = 512 floats = 2 KB scratch

  vq_main<<<NBLK, 256, 0, stream>>>(z, emb, outq, partial);
  vq_loss<<<1, 256, 0, stream>>>(partial, outq);
}
